// CausalSelfAttention_26388279066972
// MI455X (gfx1250) — matrix operand order repaired and checked by reading, not compiled, not run on hardware
//
#include <hip/hip_runtime.h>
#include <hip/hip_bf16.h>
#include <math.h>

typedef __attribute__((ext_vector_type(16))) __bf16 v16bf;
typedef __attribute__((ext_vector_type(8)))  __bf16 v8bf;
typedef __attribute__((ext_vector_type(8)))  float  v8f;
typedef __attribute__((ext_vector_type(4)))  unsigned int u32x4;
typedef __attribute__((ext_vector_type(4)))  int i32x4;
typedef __attribute__((ext_vector_type(8)))  int i32x8;

#define DEV __device__ __forceinline__

#if __has_include(<hip/amd_detail/amd_gfx1250_TDM.h>)
#define TDM_6ARG 1
#else
#define TDM_6ARG 0
#endif

// padded LDS row stride for 64-elem bf16 rows: 144 B = 36 dwords.
// lane*36 mod 64 is distinct for 16 lanes -> near-conflict-free ds_load_b128.
#define LDSW 72

DEV __bf16 f2bf(float f) {
  unsigned int u = __builtin_bit_cast(unsigned int, f);
  u += 0x7FFFu + ((u >> 16) & 1u);                 // round-to-nearest-even
  unsigned short s = (unsigned short)(u >> 16);
  return __builtin_bit_cast(__bf16, s);
}

DEV v8bf ld8(const __bf16* p) { return *(const v8bf*)p; }

DEV v16bf cat8(v8bf lo, v8bf hi) {
  return __builtin_shufflevector(lo, hi, 0,1,2,3,4,5,6,7,8,9,10,11,12,13,14,15);
}

// A fragment (16 x 32 bf16, MxK). base = ptr to row 0 of the 16-row tile.
// lane half h = lane>>4, row m = lane&15.
// elems 0..7 : K = koff + h*8 + e        elems 8..15: K = koff + 16 + h*8 + e
DEV v16bf frag_a(const __bf16* base, int stride, int koff, int lane) {
  const __bf16* r = base + (lane & 15) * stride + koff + ((lane >> 4) << 3);
  return cat8(ld8(r), ld8(r + 16));
}

// B fragment (32 x 16 bf16, KxN) from [n][k]-major storage.
// col n = lane&15 ; elems 0..7: K = koff + h*8 + e, elems 8..15: K = koff + 16 + h*8 + e
DEV v16bf frag_b(const __bf16* base, int stride, int koff, int lane) {
  const __bf16* r = base + (lane & 15) * stride + koff + ((lane >> 4) << 3);
  return cat8(ld8(r), ld8(r + 16));
}

DEV v8f wmma_bf16(v16bf a, v16bf b, v8f c) {
  return __builtin_amdgcn_wmma_f32_16x16x32_bf16(false, a, false, b, (short)0, c,
                                                 false, false);
}

DEV float hmax16(float x) {
  #pragma unroll
  for (int m = 1; m <= 8; m <<= 1) x = fmaxf(x, __shfl_xor(x, m, 32));
  return x;
}
DEV float hsum16(float x) {
  #pragma unroll
  for (int m = 1; m <= 8; m <<= 1) x += __shfl_xor(x, m, 32);
  return x;
}

// ---------------------------------------------------------------- TDM
// 2D tile load, bf16: tile0 = fastest dim (64 elems = 128 B rows), tile1 rows of
// stride0 elements. LDS padding: 4 dwords after every 32 dwords -> row stride 72.
//  g0: [1:0]=count=1 | [63:32]=lds_addr | [120:64]=global_addr | [127:126]=type=2
//  g1: [17:16]=data_size(1=2B) | [20]=pad_en | [24:22]=pad_interval |
//      [31:25]=pad_amount | [79:48]=tensor_dim0 | [111:80]=tensor_dim1 |
//      [127:112]=tile_dim0 | [143:128]=tile_dim1 | [207:160]=tensor_dim0_stride
DEV void tdm_load_2d_bf16(void* ldsdst, const __bf16* gptr,
                          unsigned tile0, unsigned tile1, unsigned stride0) {
  unsigned long long ga = (unsigned long long)(uintptr_t)gptr;
  unsigned lds_off = (unsigned)(uintptr_t)ldsdst;   // LDS aperture: low 32 bits
  u32x4 g0;
  g0[0] = 1u;                                        // count=1, user mode
  g0[1] = lds_off;
  g0[2] = (unsigned)ga;                              // global_addr[31:0]
  g0[3] = (unsigned)(ga >> 32) | (2u << 30);         // global_addr[56:32], type=2
  i32x8 g1;
  g1[0] = (int)((1u << 16)                           // data_size = 2 bytes
              | (1u << 20)                           // pad_enable
              | (4u << 22)                           // pad_interval: every 32 dw
              | (3u << 25));                         // pad_amount: 4 dwords
  g1[1] = (int)((tile0 & 0xFFFFu) << 16);            // tensor_dim0[15:0]
  g1[2] = (int)((tile0 >> 16) | ((tile1 & 0xFFFFu) << 16)); // dim0 hi | dim1 lo
  g1[3] = (int)((tile1 >> 16) | (tile0 << 16));      // dim1 hi | tile_dim0
  g1[4] = (int)tile1;                                // tile_dim1 (tile_dim2=0)
  g1[5] = (int)stride0;                              // tensor_dim0_stride[31:0]
  g1[6] = 0;
  g1[7] = 0;
  i32x4 z4 = {0, 0, 0, 0};
#if TDM_6ARG
  i32x8 z8 = {0, 0, 0, 0, 0, 0, 0, 0};
  __builtin_amdgcn_tensor_load_to_lds(g0, g1, z4, z4, z8, 0);
#else
  __builtin_amdgcn_tensor_load_to_lds(g0, g1, z4, z4, 0);
#endif
}

// ---------------------------------------------------------------- converts
__global__ void cvt_copy(const float* __restrict__ src, __bf16* __restrict__ dst, int n) {
  int i = blockIdx.x * 256 + threadIdx.x;
  if (i < n) dst[i] = f2bf(src[i]);
}

// src[rows][cols] (row-major f32) -> dst[cols][rows] (bf16)
__global__ void cvt_transpose(const float* __restrict__ src, __bf16* __restrict__ dst,
                              int rows, int cols) {
  int i = blockIdx.x * 256 + threadIdx.x;
  if (i < rows * cols) {
    int r = i / cols, c = i - r * cols;
    dst[(size_t)c * rows + r] = f2bf(src[i]);
  }
}

// ---------------------------------------------------------------- GEMM
// C(MxN) = A(MxK) * Bt(NxK)^T. 128x128 block tile, K-step 64, 3-stage TDM pipe.
// 8 waves: wave (wr,wc) computes 32x64 via 2x4 WMMA accumulators.
// mode 0: f32 C.  mode 1: scatter QKV into Q,K (B,H,L,dh) and Vt (B,H,dh,L).
__global__ __launch_bounds__(256) void gemm_bf16_128x128(
    const __bf16* __restrict__ A, const __bf16* __restrict__ Bt,
    int M, int N, int K, int mode,
    float* __restrict__ Cf,
    __bf16* __restrict__ Qb, __bf16* __restrict__ Kb, __bf16* __restrict__ Vt) {
  __shared__ __bf16 As[3][128 * LDSW];   // 18 KB per stage
  __shared__ __bf16 Bs[3][128 * LDSW];

  const int t = threadIdx.x;
  const int lane = t & 31;
  const int w = t >> 5;
  const int wr = w & 3;
  const int wc = w >> 2;
  const int bm = blockIdx.x % (M >> 7);
  const int bn = blockIdx.x / (M >> 7);
  const int m0 = bm << 7, n0 = bn << 7;

  v8f zero = {};
  v8f acc[2][4];
  #pragma unroll
  for (int i = 0; i < 2; ++i)
    #pragma unroll
    for (int j = 0; j < 4; ++j) acc[i][j] = zero;

  const __bf16* Abase = A  + (size_t)m0 * K;
  const __bf16* Bbase = Bt + (size_t)n0 * K;
  const int nstage = K >> 6;

  if (w == 0) {  // prologue: stages 0,1 in flight
    tdm_load_2d_bf16(&As[0][0], Abase, 64, 128, K);
    tdm_load_2d_bf16(&Bs[0][0], Bbase, 64, 128, K);
    if (nstage > 1) {
      tdm_load_2d_bf16(&As[1][0], Abase + 64, 64, 128, K);
      tdm_load_2d_bf16(&Bs[1][0], Bbase + 64, 64, 128, K);
    }
  }

  for (int s = 0; s < nstage; ++s) {
    const int cb = s % 3;
    if (w == 0) {
      if (s + 2 < nstage) {
        const int fb = (s + 2) % 3;    // buffer last read at stage s-1
        tdm_load_2d_bf16(&As[fb][0], Abase + (s + 2) * 64, 64, 128, K);
        tdm_load_2d_bf16(&Bs[fb][0], Bbase + (s + 2) * 64, 64, 128, K);
        __builtin_amdgcn_s_wait_tensorcnt(4);   // stage s done, s+1/s+2 in flight
      } else if (s + 1 < nstage) {
        __builtin_amdgcn_s_wait_tensorcnt(2);
      } else {
        __builtin_amdgcn_s_wait_tensorcnt(0);
      }
    }
    __syncthreads();

    #pragma unroll
    for (int ks = 0; ks < 2; ++ks) {
      v16bf a0 = frag_a(&As[cb][(wr * 32 + 0)  * LDSW], LDSW, ks * 32, lane);
      v16bf a1 = frag_a(&As[cb][(wr * 32 + 16) * LDSW], LDSW, ks * 32, lane);
      #pragma unroll
      for (int j = 0; j < 4; ++j) {
        v16bf bf = frag_b(&Bs[cb][(wc * 64 + j * 16) * LDSW], LDSW, ks * 32, lane);
        acc[0][j] = wmma_bf16(a0, bf, acc[0][j]);
        acc[1][j] = wmma_bf16(a1, bf, acc[1][j]);
      }
    }
    __syncthreads();   // reads of buffer cb done before it is refilled
  }

  // epilogue: C layout row m = (lane>>4)*8 + v, col n = lane&15
  const int lh8 = (lane >> 4) * 8;
  const int ln  = lane & 15;
  #pragma unroll
  for (int i = 0; i < 2; ++i) {
    #pragma unroll
    for (int j = 0; j < 4; ++j) {
      const int mb = m0 + wr * 32 + i * 16 + lh8;
      const int ng = n0 + wc * 64 + j * 16 + ln;
      #pragma unroll
      for (int v = 0; v < 8; ++v) {
        float val = acc[i][j][v];
        int mg = mb + v;
        if (mode == 0) {
          Cf[(size_t)mg * N + ng] = val;
        } else {
          int b = mg >> 11, l = mg & 2047;
          int sel = ng >> 10, d = ng & 1023;
          int head = d >> 6, di = d & 63;
          size_t bh = (size_t)(b * 16 + head);
          __bf16 bv = f2bf(val);
          if (sel == 0)      Qb[(bh * 2048 + l) * 64 + di] = bv;
          else if (sel == 1) Kb[(bh * 2048 + l) * 64 + di] = bv;
          else               Vt[(bh * 64 + di) * 2048 + l] = bv;
        }
      }
    }
  }
}

// ---------------------------------------------------------------- flash attention
// grid: b * H * (L/128); 8 waves, wave w owns q rows [q0, q0+16). TDM-staged K/V.
__global__ __launch_bounds__(256) void attn_fwd(
    const __bf16* __restrict__ Qb, const __bf16* __restrict__ Kb,
    const __bf16* __restrict__ Vt, __bf16* __restrict__ Ob) {
  __shared__ __bf16 Ks[2][64 * LDSW];    // [key][d]  == B [n][k] for S=Q K^T
  __shared__ __bf16 Vs[2][64 * LDSW];    // [d][key]  == B [n][k] for O=P V
  __shared__ __bf16 Ps[8][16 * LDSW];    // per-wave P tile, [m][key]

  const int t = threadIdx.x, lane = t & 31, w = t >> 5;
  const int qt = blockIdx.x & 15;
  const int h  = (blockIdx.x >> 4) & 15;
  const int b  = blockIdx.x >> 8;
  const int bh = b * 16 + h;
  const int q0 = qt * 128 + w * 16;

  const int ln  = lane & 15;
  const int lh8 = (lane >> 4) * 8;

  // Q fragments straight from global (rows are 128B aligned)
  const __bf16* qrow = Qb + ((size_t)bh * 2048 + q0 + ln) * 64;
  v16bf qf[2];
  #pragma unroll
  for (int ks = 0; ks < 2; ++ks) {
    const __bf16* r = qrow + ks * 32 + lh8;
    qf[ks] = cat8(ld8(r), ld8(r + 16));
  }

  v8f zero = {};
  v8f oacc[4];
  #pragma unroll
  for (int dj = 0; dj < 4; ++dj) oacc[dj] = zero;
  float mrow[8], lsum[8];
  #pragma unroll
  for (int v = 0; v < 8; ++v) { mrow[v] = -__builtin_inff(); lsum[v] = 0.0f; }

  const float scale = 0.125f;                      // 1/sqrt(64)
  const __bf16* kgbase = Kb + (size_t)bh * 2048 * 64;   // [key][d], stride 64
  const __bf16* vgbase = Vt + (size_t)bh * 64 * 2048;   // [d][key], stride 2048
  const int nkt = qt * 2 + 2;                      // causal tile count

  if (w == 0) {   // prologue: tile 0 -> buffer 0
    tdm_load_2d_bf16(&Ks[0][0], kgbase, 64, 64, 64);
    tdm_load_2d_bf16(&Vs[0][0], vgbase, 64, 64, 2048);
  }

  for (int kt = 0; kt < nkt; ++kt) {
    const int cb = kt & 1, nb = cb ^ 1;
    if (w == 0) {
      if (kt + 1 < nkt) {
        tdm_load_2d_bf16(&Ks[nb][0], kgbase + (size_t)(kt + 1) * 64 * 64, 64, 64, 64);
        tdm_load_2d_bf16(&Vs[nb][0], vgbase + (kt + 1) * 64, 64, 64, 2048);
        __builtin_amdgcn_s_wait_tensorcnt(2);
      } else {
        __builtin_amdgcn_s_wait_tensorcnt(0);
      }
    }
    __syncthreads();

    // S = Q K^T : 16 x 64
    v8f sacc[4];
    #pragma unroll
    for (int j = 0; j < 4; ++j) {
      sacc[j] = zero;
      #pragma unroll
      for (int ks = 0; ks < 2; ++ks) {
        v16bf bf = frag_b(&Ks[cb][(j * 16) * LDSW], LDSW, ks * 32, lane);
        sacc[j] = wmma_bf16(qf[ks], bf, sacc[j]);
      }
    }

    // scale + causal mask + row max
    float rmax[8];
    #pragma unroll
    for (int v = 0; v < 8; ++v) rmax[v] = -__builtin_inff();
    #pragma unroll
    for (int j = 0; j < 4; ++j) {
      #pragma unroll
      for (int v = 0; v < 8; ++v) {
        int qg = q0 + lh8 + v;
        int kg = kt * 64 + j * 16 + ln;
        float s = (kg > qg) ? -__builtin_inff() : sacc[j][v] * scale;
        sacc[j][v] = s;
        rmax[v] = fmaxf(rmax[v], s);
      }
    }
    float alpha[8], rsum[8];
    #pragma unroll
    for (int v = 0; v < 8; ++v) {
      rmax[v] = hmax16(rmax[v]);
      float mn = fmaxf(mrow[v], rmax[v]);
      alpha[v] = __expf(mrow[v] - mn);
      mrow[v] = mn;
      rsum[v] = 0.0f;
    }

    // P = exp(S - m) -> per-wave LDS tile (A-operand layout source)
    #pragma unroll
    for (int j = 0; j < 4; ++j) {
      #pragma unroll
      for (int v = 0; v < 8; ++v) {
        float p = __expf(sacc[j][v] - mrow[v]);
        rsum[v] += p;
        Ps[w][(lh8 + v) * LDSW + j * 16 + ln] = f2bf(p);
      }
    }
    #pragma unroll
    for (int v = 0; v < 8; ++v) {
      rsum[v] = hsum16(rsum[v]);
      lsum[v] = lsum[v] * alpha[v] + rsum[v];
    }
    #pragma unroll
    for (int dj = 0; dj < 4; ++dj)
      #pragma unroll
      for (int v = 0; v < 8; ++v) oacc[dj][v] *= alpha[v];

    // same-wave DS RAW: P stores above feed frag_a reads below
    asm volatile("s_wait_dscnt 0" ::: "memory");

    // O += P V
    v16bf pa[2];
    #pragma unroll
    for (int ks = 0; ks < 2; ++ks) pa[ks] = frag_a(&Ps[w][0], LDSW, ks * 32, lane);
    #pragma unroll
    for (int dj = 0; dj < 4; ++dj) {
      #pragma unroll
      for (int ks = 0; ks < 2; ++ks) {
        v16bf vb = frag_b(&Vs[cb][(dj * 16) * LDSW], LDSW, ks * 32, lane);
        oacc[dj] = wmma_bf16(pa[ks], vb, oacc[dj]);
      }
    }
    __syncthreads();   // buffer cb reads done before wave0 refills it
  }

  // normalize + scatter to (B, L, H*dh) bf16 for the output GEMM
  #pragma unroll
  for (int dj = 0; dj < 4; ++dj) {
    #pragma unroll
    for (int v = 0; v < 8; ++v) {
      int qg = q0 + lh8 + v;
      float o = oacc[dj][v] / lsum[v];
      Ob[((size_t)b * 2048 + qg) * 1024 + h * 64 + dj * 16 + ln] = f2bf(o);
    }
  }
}

// ---------------------------------------------------------------- launch
extern "C" void kernel_launch(void* const* d_in, const int* in_sizes, int n_in,
                              void* d_out, int out_size, void* d_ws, size_t ws_size,
                              hipStream_t stream) {
  const float* x     = (const float*)d_in[0];   // (2,2048,1024)
  const float* w_qkv = (const float*)d_in[1];   // (1024,3072)
  const float* w_out = (const float*)d_in[2];   // (1024,1024)
  float* out = (float*)d_out;

  char* ws = (char*)d_ws;
  __bf16* xb    = (__bf16*)(ws);                         // 8 MB  (4096x1024)
  __bf16* wqkvt = (__bf16*)(ws + 8388608);               // 6 MB  (3072x1024, [n][k])
  __bf16* woutt = (__bf16*)(ws + 14680064);              // 2 MB  (1024x1024, [n][k])
  __bf16* Qb    = (__bf16*)(ws + 16777216);              // 8 MB  (B,H,L,dh)
  __bf16* Kb    = (__bf16*)(ws + 25165824);              // 8 MB  (B,H,L,dh)
  __bf16* Vt    = (__bf16*)(ws + 33554432);              // 8 MB  (B,H,dh,L)
  __bf16* ao    = (__bf16*)(ws + 41943040);              // 8 MB  (B,L,D)

  cvt_copy<<<4194304 / 256, 256, 0, stream>>>(x, xb, 4194304);
  cvt_transpose<<<(1024 * 3072) / 256, 256, 0, stream>>>(w_qkv, wqkvt, 1024, 3072);
  cvt_transpose<<<(1024 * 1024) / 256, 256, 0, stream>>>(w_out, woutt, 1024, 1024);

  gemm_bf16_128x128<<<(4096 / 128) * (3072 / 128), 256, 0, stream>>>(
      xb, wqkvt, 4096, 3072, 1024, 1, nullptr, Qb, Kb, Vt);

  attn_fwd<<<2 * 16 * 16, 256, 0, stream>>>(Qb, Kb, Vt, ao);

  gemm_bf16_128x128<<<(4096 / 128) * (1024 / 128), 256, 0, stream>>>(
      ao, woutt, 4096, 1024, 1024, 0, out, nullptr, nullptr, nullptr);
}
